// DynamicTokenBasisAssociativeLM_34368328303143
// MI455X (gfx1250) — compile-verified, matrix operand order checked by reading, not run on hardware
//
#include <hip/hip_runtime.h>
#include <hip/hip_bf16.h>

// ---------------- model dims (fixed by the reference) ----------------
#define DV 32000
#define DE 512
#define DH 768
#define DM 256
#define DR 64
#define DG 16
#define DB 2
#define DT 1024
#define NTOK (DB * DT)   // 2048

typedef __attribute__((ext_vector_type(16))) __bf16 bf16x16;
typedef __attribute__((ext_vector_type(8)))  __bf16 bf16x8;
typedef __attribute__((ext_vector_type(8)))  float  f32x8;

__device__ __forceinline__ unsigned short f2bf(float f) {
    unsigned int u = __float_as_uint(f);
    u += 0x7FFFu + ((u >> 16) & 1u);          // round-to-nearest-even
    return (unsigned short)(u >> 16);
}
__device__ __forceinline__ float sigm(float x) { return 1.0f / (1.0f + expf(-x)); }

// ---------------------------------------------------------------------
// Generic fp32 -> bf16 conversion
// ---------------------------------------------------------------------
__global__ void cvt_bf16_kernel(const float* __restrict__ src,
                                unsigned short* __restrict__ dst, int n) {
    int i = blockIdx.x * 256 + threadIdx.x;
    if (i < n) dst[i] = f2bf(src[i]);
}

// ---------------------------------------------------------------------
// Embedding gather -> bf16 x (NTOK x E)
// ---------------------------------------------------------------------
__global__ void gather_x_kernel(const int* __restrict__ ids,
                                const float* __restrict__ emb,
                                unsigned short* __restrict__ x_bf) {
    int i = blockIdx.x * 256 + threadIdx.x;      // NTOK*DE = 1,048,576
    int tok = i >> 9, e = i & (DE - 1);
    x_bf[i] = f2bf(emb[(size_t)ids[tok] * DE + e]);
}

// ---------------------------------------------------------------------
// bf16 WMMA GEMM:  C(MxN) = alpha * A(MxK) @ B(NxK)^T + bias
// A,B row-major bf16 (K contiguous).  Block: 8 waves, tile 64M x 128N,
// wave tile 16M x 64N (4x v8f accumulators), K step 32.
// Fragment loads follow the CDNA5 16-bit 16x32 A layout: per lane two
// contiguous 16B chunks at K = kb..kb+7 and kb+16..kb+23, kb = 8*(lane>=16).
// ---------------------------------------------------------------------
__global__ __launch_bounds__(256) void gemm_bf16_kernel(
    const unsigned short* __restrict__ A, const unsigned short* __restrict__ B,
    const float* __restrict__ bias, float* __restrict__ Cf,
    unsigned short* __restrict__ Cbf, int M, int N, int K,
    float alpha, int squared_relu)
{
    const int lane = threadIdx.x & 31;
    const int wid  = threadIdx.x >> 5;
    const int m0 = blockIdx.y * 64  + (wid & 3) * 16;
    const int n0 = blockIdx.x * 128 + (wid >> 2) * 64;
    const int l15  = lane & 15;
    const int koff = (lane >> 4) << 3;           // 0 or 8

    f32x8 acc0 = {}, acc1 = {}, acc2 = {}, acc3 = {};
    const __bf16* Ab = (const __bf16*)A;
    const __bf16* Bb = (const __bf16*)B;
    const size_t arow = (size_t)(m0 + l15) * K;

    for (int k0 = 0; k0 < K; k0 += 32) {
        const __bf16* ap = Ab + arow + k0 + koff;
        bf16x8 a0 = *(const bf16x8*)ap;
        bf16x8 a1 = *(const bf16x8*)(ap + 16);
        bf16x16 af = __builtin_shufflevector(a0, a1,
            0,1,2,3,4,5,6,7,8,9,10,11,12,13,14,15);
#pragma unroll
        for (int j = 0; j < 4; ++j) {
            const __bf16* bp = Bb + (size_t)(n0 + j * 16 + l15) * K + k0 + koff;
            bf16x8 b0 = *(const bf16x8*)bp;
            bf16x8 b1 = *(const bf16x8*)(bp + 16);
            bf16x16 bfr = __builtin_shufflevector(b0, b1,
                0,1,2,3,4,5,6,7,8,9,10,11,12,13,14,15);
            f32x8& acc = (j == 0) ? acc0 : (j == 1) ? acc1 : (j == 2) ? acc2 : acc3;
            acc = __builtin_amdgcn_wmma_f32_16x16x32_bf16(
                false, af, false, bfr, (short)0, acc, false, false);
        }
    }

    const int mbase = m0 + ((lane >> 4) << 3);   // +0 / +8 per half-wave
#pragma unroll
    for (int j = 0; j < 4; ++j) {
        f32x8 acc = (j == 0) ? acc0 : (j == 1) ? acc1 : (j == 2) ? acc2 : acc3;
        const int n = n0 + j * 16 + l15;
        const float bv = bias ? bias[n] : 0.0f;
#pragma unroll
        for (int r = 0; r < 8; ++r) {
            float v = acc[r] * alpha + bv;
            if (squared_relu) { v = fmaxf(v, 0.0f); v = v * v; }
            size_t off = (size_t)(mbase + r) * N + n;
            if (Cf)  Cf[off]  = v;
            if (Cbf) Cbf[off] = f2bf(v);
        }
    }
}

// ---------------------------------------------------------------------
// GRU scan: sequential over T, single 1024-thread workgroup, fp32 exact.
// xi (NTOK x 3H) precomputed by WMMA GEMM.  h/hh staged in LDS.
// ---------------------------------------------------------------------
__global__ __launch_bounds__(1024) void gru_scan_kernel(
    const float* __restrict__ xi, const float* __restrict__ W_hh,
    const float* __restrict__ b_hh, float* __restrict__ states,
    unsigned short* __restrict__ states_bf)
{
    __shared__ float h[DB * DH];          // 1536
    __shared__ float hh[DB * 3 * DH];     // 4608
    const int tid = threadIdx.x;
    for (int i = tid; i < DB * DH; i += 1024) h[i] = 0.0f;
    __syncthreads();

    for (int t = 0; t < DT; ++t) {
        // hh = h @ W_hh^T + b_hh
        for (int idx = tid; idx < DB * 3 * DH; idx += 1024) {
            int b = idx / (3 * DH), j = idx - b * (3 * DH);
            const float* w = W_hh + (size_t)j * DH;
            const float* hb = h + b * DH;
            float s = b_hh[j];
#pragma unroll 4
            for (int k = 0; k < DH; ++k) s += hb[k] * w[k];
            hh[idx] = s;
        }
        __syncthreads();
        // gate math + state update (torch order r,z,n)
        for (int idx = tid; idx < DB * DH; idx += 1024) {
            int b = idx / DH, j = idx - b * DH;
            size_t xb = (size_t)(b * DT + t) * (3 * DH);
            float r = sigm(xi[xb + j]           + hh[b * 3 * DH + j]);
            float z = sigm(xi[xb + DH + j]      + hh[b * 3 * DH + DH + j]);
            float n = tanhf(xi[xb + 2 * DH + j] + r * hh[b * 3 * DH + 2 * DH + j]);
            float hp = (1.0f - z) * n + z * h[idx];
            h[idx] = hp;
            size_t o = (size_t)(b * DT + t) * DH + j;
            states[o] = hp;
            states_bf[o] = f2bf(hp);
        }
        __syncthreads();
    }
}

// ---------------------------------------------------------------------
// statesT[b][h][t] = states_bf[b][t][h]   (bf16, for attn@states GEMM)
// ---------------------------------------------------------------------
__global__ void transpose_states_kernel(const unsigned short* __restrict__ sbf,
                                        unsigned short* __restrict__ sT) {
    int i = blockIdx.x * 256 + threadIdx.x;      // DB*DH*DT = 1,572,864
    int b = i / (DH * DT);
    int rem = i - b * (DH * DT);
    int hh = rem / DT, t = rem - hh * DT;
    sT[i] = sbf[(size_t)(b * DT + t) * DH + hh];
}

// ---------------------------------------------------------------------
// Strict-causal masked softmax + renorm, in place on scores (B,T,T).
// Row t: valid keys s in [0, t).  Writes f32 probs (in place) + bf16 copy.
// ---------------------------------------------------------------------
__global__ __launch_bounds__(256) void softmax_causal_kernel(
    float* __restrict__ scores, unsigned short* __restrict__ attn_bf)
{
    const int row = blockIdx.x;                  // 0..2047
    const int b = row >> 10, t = row & (DT - 1);
    float* r = scores + ((size_t)b << 20) + (size_t)t * DT;
    unsigned short* o = attn_bf + ((size_t)b << 20) + (size_t)t * DT;
    const int tid = threadIdx.x;
    __shared__ float red[256];

    if (t == 0) {
        for (int s = tid; s < DT; s += 256) { r[s] = 0.0f; o[s] = 0; }
        return;
    }
    float mx = -3.4e38f;
    for (int s = tid; s < t; s += 256) mx = fmaxf(mx, r[s]);
    red[tid] = mx; __syncthreads();
    for (int off = 128; off; off >>= 1) {
        if (tid < off) red[tid] = fmaxf(red[tid], red[tid + off]);
        __syncthreads();
    }
    mx = red[0]; __syncthreads();
    float sm = 0.0f;
    for (int s = tid; s < t; s += 256) sm += expf(r[s] - mx);
    red[tid] = sm; __syncthreads();
    for (int off = 128; off; off >>= 1) {
        if (tid < off) red[tid] += red[tid + off];
        __syncthreads();
    }
    const float inv = 1.0f / fmaxf(red[0], 1e-6f);
    __syncthreads();
    for (int s = tid; s < DT; s += 256) {
        float p = (s < t) ? expf(r[s] - mx) * inv : 0.0f;
        r[s] = p;
        o[s] = f2bf(p);
    }
}

// ---------------------------------------------------------------------
// Top-2 router: one wave32 per token.
// ---------------------------------------------------------------------
__global__ __launch_bounds__(256) void route_topk_kernel(
    const float* __restrict__ states, const float* __restrict__ Wr,
    const float* __restrict__ br, float* __restrict__ route)
{
    const int lane = threadIdx.x & 31, wid = threadIdx.x >> 5;
    const int tok = blockIdx.x * 8 + wid;
    const float* st = states + (size_t)tok * DH;
    float lg = 0.0f;
    for (int g = 0; g < DG; ++g) {
        float s = 0.0f;
        for (int k = lane; k < DH; k += 32) s += st[k] * Wr[g * DH + k];
        for (int off = 16; off; off >>= 1) s += __shfl_xor(s, off, 32);
        if (lane == g) lg = s + br[g];
    }
    float tv[DG];
    for (int g = 0; g < DG; ++g) tv[g] = __shfl(lg, g, 32);
    float v1 = -3.4e38f; int i1 = 0;
    for (int g = 0; g < DG; ++g) if (tv[g] > v1) { v1 = tv[g]; i1 = g; }
    float v2 = -3.4e38f; int i2 = 0;
    for (int g = 0; g < DG; ++g) if (g != i1 && tv[g] > v2) { v2 = tv[g]; i2 = g; }
    const float e2 = expf(v2 - v1), denom = 1.0f + e2;
    if (lane < DG)
        route[tok * DG + lane] =
            (lane == i1) ? (1.0f / denom) : ((lane == i2) ? (e2 / denom) : 0.0f);
}

// ---------------------------------------------------------------------
// Routed bases (only the K=2 active groups) + gate + low-rank delta,
// fused into final_feat = bf16(base_feat + delta).  Also eg (copy gate).
// One wave32 per token.
// ---------------------------------------------------------------------
__global__ __launch_bounds__(256) void routed_delta_kernel(
    const float* __restrict__ base_feat, const float* __restrict__ mem_states,
    const float* __restrict__ states, const float* __restrict__ route,
    const float* __restrict__ base_cw, const float* __restrict__ base_cb,
    const float* __restrict__ mem_cw, const float* __restrict__ mem_cb,
    const float* __restrict__ Wbg, const float* __restrict__ bbg,
    const float* __restrict__ Wbu, const float* __restrict__ Wg,
    const float* __restrict__ bg, const float* __restrict__ basis_scale,
    const float* __restrict__ exact_scale,
    unsigned short* __restrict__ final_feat_bf, float* __restrict__ eg)
{
    const int lane = threadIdx.x & 31, wid = threadIdx.x >> 5;
    const int tok = blockIdx.x * 8 + wid;
    __shared__ float comb[8][DR];

    const float* bfp = base_feat + (size_t)tok * DE;
    const float* ms  = mem_states + (size_t)tok * DH;
    const float* st  = states + (size_t)tok * DH;

    float accb[2] = {0.0f, 0.0f}, accm[2] = {0.0f, 0.0f};
    for (int g = 0; g < DG; ++g) {
        const float w = route[tok * DG + g];
        if (w <= 0.0f) continue;
#pragma unroll
        for (int r2 = 0; r2 < 2; ++r2) {
            const int r = lane + 32 * r2;
            const float* cw = base_cw + ((size_t)g * DE) * DR + r;
            float s = base_cb[g * DR + r];
            for (int d = 0; d < DE; ++d) s += bfp[d] * cw[(size_t)d * DR];
            accb[r2] += w * s;
            const float* mw = mem_cw + ((size_t)g * DH) * DR + r;
            float s2 = mem_cb[g * DR + r];
            for (int d = 0; d < DH; ++d) s2 += ms[d] * mw[(size_t)d * DR];
            accm[r2] += w * s2;
        }
    }
    const float bs = basis_scale[0];
#pragma unroll
    for (int r2 = 0; r2 < 2; ++r2) {
        const int r = lane + 32 * r2;
        float gs = bbg[r];
        for (int d = 0; d < DH; ++d) gs += st[d] * Wbg[(size_t)r * DH + d];
        comb[wid][r] = accb[r2] + bs * sigm(gs) * accm[r2];
    }
    // copy-gate eg = sigmoid(states @ Wg^T + bg) * exact_scale
    float es = 0.0f;
    for (int d = lane; d < DH; d += 32) es += st[d] * Wg[d];
    for (int off = 16; off; off >>= 1) es += __shfl_xor(es, off, 32);
    if (lane == 0) eg[tok] = sigm(es + bg[0]) * exact_scale[0];
    __syncthreads();
    // delta = comb @ Wbu^T ; final_feat = bf16(base_feat + delta)
    for (int e = lane; e < DE; e += 32) {
        float d = 0.0f;
#pragma unroll 4
        for (int r = 0; r < DR; ++r) d += comb[wid][r] * Wbu[(size_t)e * DR + r];
        final_feat_bf[(size_t)tok * DE + e] = f2bf(bfp[e] + d);
    }
}

// ---------------------------------------------------------------------
// Copy mechanism: logits[b,t, ids[b,s]] += attn[b,t,s] * eg[b,t]
// ---------------------------------------------------------------------
__global__ void copy_scatter_kernel(const float* __restrict__ attn,
                                    const int* __restrict__ ids,
                                    const float* __restrict__ eg,
                                    float* __restrict__ out)
{
    size_t idx = (size_t)blockIdx.x * 256 + threadIdx.x;   // b*2^20 + t*1024 + s
    const int b = (int)(idx >> 20);
    const int t = (int)((idx >> 10) & (DT - 1));
    const int s = (int)(idx & (DT - 1));
    if (s >= t) return;
    const float a = attn[idx];
    if (a != 0.0f) {
        const int v = ids[(b << 10) + s];
        atomicAdd(out + (size_t)((b << 10) + t) * DV + v, a * eg[(b << 10) + t]);
    }
}

// ---------------------------------------------------------------------
// host side
// ---------------------------------------------------------------------
static void gemm(hipStream_t s, const void* A, const void* B, const float* bias,
                 float* Cf, void* Cbf, int M, int N, int K,
                 float alpha = 1.0f, int sqrelu = 0) {
    dim3 g(N / 128, M / 64);
    gemm_bf16_kernel<<<g, 256, 0, s>>>((const unsigned short*)A,
        (const unsigned short*)B, bias, Cf, (unsigned short*)Cbf,
        M, N, K, alpha, sqrelu);
}
static void cvt(hipStream_t s, const float* src, void* dst, int n) {
    cvt_bf16_kernel<<<(n + 255) / 256, 256, 0, s>>>(src, (unsigned short*)dst, n);
}

extern "C" void kernel_launch(void* const* d_in, const int* in_sizes, int n_in,
                              void* d_out, int out_size, void* d_ws, size_t ws_size,
                              hipStream_t stream) {
    (void)in_sizes; (void)n_in; (void)out_size; (void)ws_size;

    const int*   ids    = (const int*)  d_in[0];
    const float* emb    = (const float*)d_in[1];
    const float* W_ih   = (const float*)d_in[2];
    // W_hh f32 used directly in the scan
    const float* W_hh   = (const float*)d_in[3];
    const float* b_ih   = (const float*)d_in[4];
    const float* b_hh   = (const float*)d_in[5];
    const float* Wq     = (const float*)d_in[6];
    const float* bq     = (const float*)d_in[7];
    const float* Wk     = (const float*)d_in[8];
    const float* bk     = (const float*)d_in[9];
    const float* Wg     = (const float*)d_in[10];
    const float* bg     = (const float*)d_in[11];
    const float* Whf    = (const float*)d_in[12];
    const float* bhf    = (const float*)d_in[13];
    const float* Whp    = (const float*)d_in[14];
    const float* bhp    = (const float*)d_in[15];
    const float* Wr     = (const float*)d_in[16];
    const float* br     = (const float*)d_in[17];
    const float* base_cw = (const float*)d_in[18];
    const float* base_cb = (const float*)d_in[19];
    const float* mem_cw  = (const float*)d_in[20];
    const float* mem_cb  = (const float*)d_in[21];
    const float* Wbg    = (const float*)d_in[22];
    const float* bbg    = (const float*)d_in[23];
    const float* Wbu    = (const float*)d_in[24];
    const float* out_bias = (const float*)d_in[25];
    const float* exact_scale = (const float*)d_in[26];
    const float* basis_scale = (const float*)d_in[27];
    float* out = (float*)d_out;

    // ---- workspace bump allocator ----
    char* wp = (char*)d_ws;
    auto alloc = [&](size_t bytes) -> void* {
        void* r = (void*)wp;
        wp += (bytes + 255) & ~(size_t)255;
        return r;
    };
    void* emb_bf   = alloc((size_t)DV * DE * 2);
    void* x_bf     = alloc((size_t)NTOK * DE * 2);
    void* Wih_bf   = alloc((size_t)3 * DH * DE * 2);
    void* Whf_bf   = alloc((size_t)4 * DE * DH * 2);
    void* Whp_bf   = alloc((size_t)DE * 4 * DE * 2);
    void* Wq_bf    = alloc((size_t)DM * DH * 2);
    void* Wk_bf    = alloc((size_t)DM * DH * 2);
    float* xi      = (float*)alloc((size_t)NTOK * 3 * DH * 4);
    float* states  = (float*)alloc((size_t)NTOK * DH * 4);
    void* states_bf  = alloc((size_t)NTOK * DH * 2);
    void* statesT_bf = alloc((size_t)NTOK * DH * 2);
    void* head_bf  = alloc((size_t)NTOK * 4 * DE * 2);
    float* base_feat = (float*)alloc((size_t)NTOK * DE * 4);
    void* q_bf     = alloc((size_t)NTOK * DM * 2);
    void* k_bf     = alloc((size_t)NTOK * DM * 2);
    float* scores  = (float*)alloc((size_t)DB * DT * DT * 4);   // becomes attn f32
    void* attn_bf  = alloc((size_t)DB * DT * DT * 2);
    float* mem_states = (float*)alloc((size_t)NTOK * DH * 4);
    float* route   = (float*)alloc((size_t)NTOK * DG * 4);
    float* eg      = (float*)alloc((size_t)NTOK * 4);
    void* final_bf = alloc((size_t)NTOK * DE * 2);

    // ---- weight conversions (bf16 for WMMA) ----
    cvt(stream, emb, emb_bf, DV * DE);
    cvt(stream, W_ih, Wih_bf, 3 * DH * DE);
    cvt(stream, Whf, Whf_bf, 4 * DE * DH);
    cvt(stream, Whp, Whp_bf, DE * 4 * DE);
    cvt(stream, Wq, Wq_bf, DM * DH);
    cvt(stream, Wk, Wk_bf, DM * DH);

    // ---- embedding gather ----
    gather_x_kernel<<<(NTOK * DE) / 256, 256, 0, stream>>>(ids, emb, (unsigned short*)x_bf);

    // ---- GRU: xi = x @ W_ih^T + b_ih (WMMA), then sequential scan ----
    gemm(stream, x_bf, Wih_bf, b_ih, xi, nullptr, NTOK, 3 * DH, DE);
    gru_scan_kernel<<<1, 1024, 0, stream>>>(xi, W_hh, b_hh, states,
                                            (unsigned short*)states_bf);

    // ---- squared-relu head -> base_feat ----
    gemm(stream, states_bf, Whf_bf, bhf, nullptr, head_bf, NTOK, 4 * DE, DH,
         1.0f, /*squared_relu=*/1);
    gemm(stream, head_bf, Whp_bf, bhp, base_feat, nullptr, NTOK, DE, 4 * DE);

    // ---- attention: q/k, scores, masked softmax, attn @ states ----
    gemm(stream, states_bf, Wq_bf, bq, nullptr, q_bf, NTOK, DM, DH);
    gemm(stream, states_bf, Wk_bf, bk, nullptr, k_bf, NTOK, DM, DH);
    const float inv_sqrt_m = 1.0f / 16.0f;   // 1/sqrt(256)
    for (int b = 0; b < DB; ++b)
        gemm(stream,
             (char*)q_bf + (size_t)b * DT * DM * 2,
             (char*)k_bf + (size_t)b * DT * DM * 2,
             nullptr, scores + (size_t)b * DT * DT, nullptr,
             DT, DT, DM, inv_sqrt_m);
    softmax_causal_kernel<<<NTOK, 256, 0, stream>>>(scores, (unsigned short*)attn_bf);
    transpose_states_kernel<<<(DB * DH * DT) / 256, 256, 0, stream>>>(
        (const unsigned short*)states_bf, (unsigned short*)statesT_bf);
    for (int b = 0; b < DB; ++b)
        gemm(stream,
             (char*)attn_bf + (size_t)b * DT * DT * 2,
             (char*)statesT_bf + (size_t)b * DH * DT * 2,
             nullptr, mem_states + (size_t)b * DT * DH, nullptr,
             DT, DH, DT);

    // ---- router + routed bases + delta -> final_feat ----
    route_topk_kernel<<<NTOK / 8, 256, 0, stream>>>(states, Wr, br, route);
    routed_delta_kernel<<<NTOK / 8, 256, 0, stream>>>(
        base_feat, mem_states, states, route, base_cw, base_cb, mem_cw, mem_cb,
        Wbg, bbg, Wbu, Wg, bg, basis_scale, exact_scale,
        (unsigned short*)final_bf, eg);

    // ---- vocab logits: (base_feat + delta) @ emb^T + out_bias ----
    gemm(stream, final_bf, emb_bf, out_bias, out, nullptr, NTOK, DV, DE);

    // ---- copy mechanism scatter-add ----
    copy_scatter_kernel<<<(DB * DT * DT) / 256, 256, 0, stream>>>(
        scores, ids, eg, out);
}
